// LinearAttention_28879360098404
// MI455X (gfx1250) — matrix operand order repaired and checked by reading, not compiled, not run on hardware
//
#include <hip/hip_runtime.h>

#define DEV __device__ __forceinline__

typedef __attribute__((ext_vector_type(16))) __bf16 v16bf;
typedef __attribute__((ext_vector_type(8)))  float  v8f;

constexpr int Bn = 2, Sn = 1024, Dn = 256, Hn = 8, Kn = 32;
constexpr int BS = Bn * Sn;                    // 2048 rows
constexpr int CHUNK = 64, NCHUNK = Sn / CHUNK; // 16 chunks per (h,b)
constexpr float EPSI = 1e-9f;

// ---------- bf16 helpers ----------
DEV __bf16 f2bf(float f) {
  union { float f; unsigned u; } a; a.f = f;
  unsigned r = a.u + 0x7FFFu + ((a.u >> 16) & 1u);   // round-to-nearest-even
  unsigned short h = (unsigned short)(r >> 16);
  __bf16 b; __builtin_memcpy(&b, &h, 2); return b;
}
DEV float bf2f(__bf16 b) {
  unsigned short h; __builtin_memcpy(&h, &b, 2);
  union { unsigned u; float f; } a; a.u = ((unsigned)h) << 16;
  return a.f;
}

// ---------- WMMA wrapper: D = A(16x32 bf16) * B(32x16 bf16) + C(16x16 f32) ----------
DEV v8f wmma_bf16(v16bf a, v16bf b, v8f c) {
  return __builtin_amdgcn_wmma_f32_16x16x32_bf16(
      false, a, false, b, (short)0, c, false, false);
}

// ---------- fragment loaders (contiguous k = 16*(lane/16) + s in A and B, wave32) ----------
// All loaders below read CONTIGUOUS 16-byte runs per lane -> global/ds b128.
// A fragment: A[m][kk] = p[m*ld + kk]
DEV v16bf load_a(const __bf16* p, int ld, int lane) {
  int m = lane & 15, g = lane >> 4;
  const __bf16* row = p + (size_t)m * ld;
  v16bf a;
#pragma unroll
  for (int j = 0; j < 8; ++j) {
    int kk = g * 16 + j * 2;
    a[2 * j]     = row[kk];
    a[2 * j + 1] = row[kk + 1];
  }
  return a;
}
// B fragment from K-contiguous (transposed) source: B[kk][n] = p[n*ld + kk]
DEV v16bf load_bt(const __bf16* p, int ld, int lane) {
  int n = lane & 15, kb = (lane >> 4) * 16;
  const __bf16* row = p + (size_t)n * ld;
  v16bf b;
#pragma unroll
  for (int j = 0; j < 8; ++j) {
    b[2 * j]     = row[kb + 2 * j];
    b[2 * j + 1] = row[kb + 2 * j + 1];
  }
  return b;
}

// =====================================================================
// Kernel 0a: fp32 -> bf16 conversion (x)
__global__ void k_cvt_bf16(const float* __restrict__ s, __bf16* __restrict__ d, int n) {
  int i = blockIdx.x * blockDim.x + threadIdx.x;
  if (i < n) d[i] = f2bf(s[i]);
}
// Kernel 0b: fp32 -> bf16 + per-head transpose: [H][R][C] -> [H][C][R]
__global__ void k_cvt_tr(const float* __restrict__ s, __bf16* __restrict__ d,
                         int R, int C, int total) {
  int i = blockIdx.x * blockDim.x + threadIdx.x;
  if (i >= total) return;
  int h = i / (R * C), rem = i % (R * C);
  int r = rem / C, c = rem % C;
  d[((size_t)h * C + c) * R + r] = f2bf(s[i]);
}

// Kernel 1: Q/K projections + elu+1.  tasks = Hn * (BS/16) * 2
// Weights pre-transposed: Wt [H][Kn][Dn].  Also emits Kt [H][Kn][BS].
__global__ void k_proj_qk(const __bf16* __restrict__ xb,
                          const __bf16* __restrict__ Wkt,
                          const __bf16* __restrict__ Wqt,
                          __bf16* __restrict__ Qb, __bf16* __restrict__ Kb,
                          __bf16* __restrict__ Kt) {
  int lane = threadIdx.x & 31;
  int task = blockIdx.x * (blockDim.x >> 5) + (threadIdx.x >> 5);
  int h = task / ((BS / 16) * 2);
  int rem = task % ((BS / 16) * 2);
  int rt = rem >> 1, isq = rem & 1;
  const __bf16* Wt = isq ? Wqt : Wkt;
  __bf16* dst = isq ? Qb : Kb;
  v8f acc0 = {0,0,0,0,0,0,0,0}, acc1 = {0,0,0,0,0,0,0,0};
#pragma unroll
  for (int ds = 0; ds < Dn / 32; ++ds) {
    v16bf a  = load_a (xb + (size_t)(rt * 16) * Dn + ds * 32, Dn, lane);
    v16bf b0 = load_bt(Wt + ((size_t)h * Kn + 0 ) * Dn + ds * 32, Dn, lane);
    v16bf b1 = load_bt(Wt + ((size_t)h * Kn + 16) * Dn + ds * 32, Dn, lane);
    acc0 = wmma_bf16(a, b0, acc0);
    acc1 = wmma_bf16(a, b1, acc1);
  }
  int n = lane & 15, g = lane >> 4;
#pragma unroll
  for (int j = 0; j < 8; ++j) {
    int r = rt * 16 + g * 8 + j;
    float v0 = acc0[j], v1 = acc1[j];
    v0 = v0 > 0.f ? v0 + 1.f : __expf(v0);   // elu(x)+1
    v1 = v1 > 0.f ? v1 + 1.f : __expf(v1);
    __bf16 h0 = f2bf(v0), h1 = f2bf(v1);
    dst[(size_t)(h * BS + r) * Kn + n]      = h0;
    dst[(size_t)(h * BS + r) * Kn + 16 + n] = h1;
    if (!isq) {  // transposed key copy for chunk-KV GEMM (wave-uniform branch)
      Kt[((size_t)h * Kn + n)      * BS + r] = h0;
      Kt[((size_t)h * Kn + 16 + n) * BS + r] = h1;
    }
  }
}

// Kernel 2: V projection, stored TRANSPOSED: Vt [H][Dn][BS].
__global__ void k_proj_vt(const __bf16* __restrict__ xb,
                          const __bf16* __restrict__ Wvt,
                          __bf16* __restrict__ Vt) {
  int lane = threadIdx.x & 31;
  int task = blockIdx.x * (blockDim.x >> 5) + (threadIdx.x >> 5);
  int h = task / ((BS / 16) * (Dn / 16));
  int rem = task % ((BS / 16) * (Dn / 16));
  int rt = rem / (Dn / 16), nt = rem % (Dn / 16);
  v8f acc = {0,0,0,0,0,0,0,0};
#pragma unroll
  for (int ds = 0; ds < Dn / 32; ++ds) {
    __builtin_prefetch(xb + (size_t)(rt * 16) * Dn + (ds + 1) * 32, 0, 1);
    v16bf a = load_a (xb + (size_t)(rt * 16) * Dn + ds * 32, Dn, lane);
    v16bf b = load_bt(Wvt + ((size_t)h * Dn + nt * 16) * Dn + ds * 32, Dn, lane);
    acc = wmma_bf16(a, b, acc);
  }
  int n = lane & 15, g = lane >> 4;
#pragma unroll
  for (int j = 0; j < 8; ++j)
    Vt[((size_t)h * Dn + nt * 16 + n) * BS + rt * 16 + g * 8 + j] = f2bf(acc[j]);
}

// Kernel 3: cumulative key sums per (h,b); 16 blocks x 32 threads.
__global__ void k_kcum(const __bf16* __restrict__ Kb, float* __restrict__ kcum) {
  int hb = blockIdx.x, h = hb / Bn, b = hb % Bn;
  int k = threadIdx.x;
  float acc = 0.f;
  for (int s = 0; s < Sn; ++s) {
    size_t idx = (size_t)(h * BS + b * Sn + s) * Kn + k;
    acc += bf2f(Kb[idx]);
    kcum[idx] = acc;
  }
}

// Kernel 4: per-chunk KV sums: chunkKV[hb,c][d][k] = sum_t V[t,d]K[t,k]
__global__ void k_chunk_kv(const __bf16* __restrict__ Kt,
                           const __bf16* __restrict__ Vt,
                           float* __restrict__ chunkKV) {
  int lane = threadIdx.x & 31;
  int task = blockIdx.x * (blockDim.x >> 5) + (threadIdx.x >> 5);
  const int tiles = (Dn / 16) * (Kn / 16); // 32
  int cid = task / tiles, tile = task % tiles;
  int dt = tile >> 1, kt = tile & 1;
  int hb = cid / NCHUNK, c = cid % NCHUNK;
  int h = hb / Bn, b = hb % Bn;
  int rowb = b * Sn + c * CHUNK;
  v8f acc = {0,0,0,0,0,0,0,0};
#pragma unroll
  for (int st2 = 0; st2 < 2; ++st2) {
    // A[m=d][kk=t] from Vt (ld=BS, contiguous t); B[kk=t][n=k] from Kt.
    v16bf a  = load_a (Vt + ((size_t)h * Dn + dt * 16) * BS + rowb + st2 * 32, BS, lane);
    v16bf bb = load_bt(Kt + ((size_t)h * Kn + kt * 16) * BS + rowb + st2 * 32, BS, lane);
    acc = wmma_bf16(a, bb, acc);
  }
  int n = lane & 15, g = lane >> 4;
#pragma unroll
  for (int j = 0; j < 8; ++j)
    chunkKV[(size_t)cid * (Dn * Kn) + (size_t)(dt * 16 + g * 8 + j) * Kn + kt * 16 + n] = acc[j];
}

// Kernel 5: per-(h,b,chunk): node = (Q*State + causal(QK^T)*V) / den.
__global__ void k_node(const __bf16* __restrict__ Qb, const __bf16* __restrict__ Kb,
                       const __bf16* __restrict__ Vt, const float* __restrict__ kcum,
                       const float* __restrict__ chunkKV,
                       __bf16* __restrict__ nodeb, float* __restrict__ deng) {
  __shared__ __bf16 stateBf[Dn * Kn];  // 16 KB: prefix KV state (bf16, [d][k])
  __shared__ __bf16 sc[CHUNK * CHUNK]; // 8 KB: masked intra-chunk scores
  __shared__ float  denl[CHUNK];
  int tid = threadIdx.x, lane = tid & 31, w = tid >> 5;
  int hb = blockIdx.x / NCHUNK, c = blockIdx.x % NCHUNK;
  int h = hb / Bn, b = hb % Bn;
  int rowb = b * Sn + c * CHUNK;

  // phase 1: prefix state from prior chunks (f32 accum in regs, bf16 to LDS)
  for (int i = tid; i < Dn * Kn; i += blockDim.x) {
    float acc = 0.f;
    const float* p = chunkKV + (size_t)(hb * NCHUNK) * (Dn * Kn) + i;
    for (int pc = 0; pc < c; ++pc) acc += p[(size_t)pc * (Dn * Kn)];
    stateBf[i] = f2bf(acc);
  }
  // phase 2: denominators den[s] = q[s] . kcum[s]
  if (tid < CHUNK) {
    int r = rowb + tid;
    const __bf16* q = Qb + (size_t)(h * BS + r) * Kn;
    const float* kc = kcum + (size_t)(h * BS + r) * Kn;
    float acc = 0.f;
    for (int k = 0; k < Kn; ++k) acc += bf2f(q[k]) * kc[k];
    denl[tid] = acc;
    deng[h * BS + r] = acc;
  }
  __syncthreads();

  // phase 3: causal intra-chunk scores -> LDS (bf16). 16 tiles / 8 waves.
  for (int ti = w; ti < 16; ti += (int)(blockDim.x >> 5)) {
    int mt = ti >> 2, tt = ti & 3;
    int g = lane >> 4, n = lane & 15;
    if (tt <= mt) {
      v16bf a  = load_a (Qb + (size_t)(h * BS + rowb + mt * 16) * Kn, Kn, lane);
      v16bf bt = load_bt(Kb + (size_t)(h * BS + rowb + tt * 16) * Kn, Kn, lane);
      v8f z = {0,0,0,0,0,0,0,0};
      v8f s0 = wmma_bf16(a, bt, z);
#pragma unroll
      for (int j = 0; j < 8; ++j) {
        int sl = mt * 16 + g * 8 + j, tl = tt * 16 + n;
        sc[sl * CHUNK + tl] = f2bf(tl <= sl ? s0[j] : 0.f);
      }
    } else {
#pragma unroll
      for (int j = 0; j < 8; ++j)
        sc[(mt * 16 + g * 8 + j) * CHUNK + tt * 16 + n] = f2bf(0.f);
    }
  }
  __syncthreads();

  // phase 4: node tiles. 64 tiles / 8 waves, 3 WMMAs each.
  for (int ti = w; ti < 4 * (Dn / 16); ti += (int)(blockDim.x >> 5)) {
    int mt = ti / (Dn / 16), nt = ti % (Dn / 16);
    int g = lane >> 4, n = lane & 15;
    v8f acc = {0,0,0,0,0,0,0,0};
    { // inter-chunk: Q (16x32) x State^T (32x16); B from LDS, contiguous 32B/lane
      v16bf a   = load_a (Qb + (size_t)(h * BS + rowb + mt * 16) * Kn, Kn, lane);
      v16bf bfr = load_bt(stateBf + (size_t)(nt * 16) * Kn, Kn, lane);
      acc = wmma_bf16(a, bfr, acc);
    }
#pragma unroll
    for (int st2 = 0; st2 < 2; ++st2) { // intra-chunk: sc(16x64) x V(64x16)
      v16bf a = load_a (sc + (size_t)(mt * 16) * CHUNK + st2 * 32, CHUNK, lane);
      v16bf b = load_bt(Vt + ((size_t)h * Dn + nt * 16) * BS + rowb + st2 * 32, BS, lane);
      acc = wmma_bf16(a, b, acc);
    }
#pragma unroll
    for (int j = 0; j < 8; ++j) {
      int sl = mt * 16 + g * 8 + j;
      float v = acc[j] / (denl[sl] + EPSI);
      nodeb[(size_t)(h * BS + rowb + sl) * Dn + nt * 16 + n] = f2bf(v);
    }
  }
}

// Kernel 6: att map. Each wave: one 16x64 strip (1 Q frag + den reused over 4 WMMAs).
// tasks = Hn*Bn * 64 * 16
__global__ void k_att(const __bf16* __restrict__ Qb, const __bf16* __restrict__ Kb,
                      const float* __restrict__ deng, float* __restrict__ att) {
  int lane = threadIdx.x & 31;
  int task = blockIdx.x * (blockDim.x >> 5) + (threadIdx.x >> 5);
  int hb = task >> 10, rem = task & 1023;
  int st = rem >> 4, ttg = rem & 15;
  int h = hb / Bn, b = hb % Bn;
  int g = lane >> 4, n = lane & 15;
  size_t attb = (size_t)hb * Sn * Sn;
  v16bf a;
#pragma unroll
  for (int i = 0; i < 16; ++i) a[i] = f2bf(0.f);
  float dinv[8] = {};
  bool any = (ttg * 4) <= st;
  if (any) {
    a = load_a(Qb + (size_t)(h * BS + b * Sn + st * 16) * Kn, Kn, lane);
#pragma unroll
    for (int j = 0; j < 8; ++j)
      dinv[j] = 1.f / (deng[h * BS + b * Sn + st * 16 + g * 8 + j] + EPSI);
  }
#pragma unroll
  for (int c4 = 0; c4 < 4; ++c4) {
    int tt = ttg * 4 + c4;
    if (tt > st) { // fully-masked tile: write zeros (d_out is poisoned)
#pragma unroll
      for (int j = 0; j < 8; ++j)
        att[attb + (size_t)(st * 16 + g * 8 + j) * Sn + tt * 16 + n] = 0.f;
    } else {
      v16bf bt = load_bt(Kb + (size_t)(h * BS + b * Sn + tt * 16) * Kn, Kn, lane);
      v8f z = {0,0,0,0,0,0,0,0};
      v8f s0 = wmma_bf16(a, bt, z);
#pragma unroll
      for (int j = 0; j < 8; ++j) {
        int sl = st * 16 + g * 8 + j, tl = tt * 16 + n;
        float v = (tl <= sl) ? s0[j] * dinv[j] : 0.f;
        att[attb + (size_t)sl * Sn + tl] = v;
      }
    }
  }
}

// Kernel 7: out[r,e] = sum_h sum_d node[h,r,d] * Wo[h,d,e].  Wot pre-transposed.
__global__ void k_out(const __bf16* __restrict__ nodeb, const __bf16* __restrict__ Wot,
                      float* __restrict__ out) {
  int lane = threadIdx.x & 31;
  int task = blockIdx.x * (blockDim.x >> 5) + (threadIdx.x >> 5);
  int mt = task / (Dn / 16), nt = task % (Dn / 16);
  int g = lane >> 4, n = lane & 15;
  v8f acc = {0,0,0,0,0,0,0,0};
  for (int h = 0; h < Hn; ++h) {
#pragma unroll
    for (int ds = 0; ds < Dn / 32; ++ds) {
      __builtin_prefetch(nodeb + (size_t)(h * BS + mt * 16) * Dn + (ds + 1) * 32, 0, 1);
      v16bf a = load_a (nodeb + (size_t)(h * BS + mt * 16) * Dn + ds * 32, Dn, lane);
      v16bf b = load_bt(Wot + ((size_t)h * Dn + nt * 16) * Dn + ds * 32, Dn, lane);
      acc = wmma_bf16(a, b, acc);
    }
  }
#pragma unroll
  for (int j = 0; j < 8; ++j)
    out[(size_t)(mt * 16 + g * 8 + j) * Dn + nt * 16 + n] = acc[j];
}

// =====================================================================
extern "C" void kernel_launch(void* const* d_in, const int* in_sizes, int n_in,
                              void* d_out, int out_size, void* d_ws, size_t ws_size,
                              hipStream_t stream) {
  (void)in_sizes; (void)n_in; (void)out_size; (void)ws_size;
  const float* x  = (const float*)d_in[0];
  const float* Wk = (const float*)d_in[1];
  const float* Wq = (const float*)d_in[2];
  const float* Wv = (const float*)d_in[3];
  const float* Wo = (const float*)d_in[4];

  char* ws = (char*)d_ws;
  size_t off = 0;
  __bf16* xb   = (__bf16*)(ws + off); off += (size_t)BS * Dn * 2;        // 1.0 MB
  __bf16* Wkt  = (__bf16*)(ws + off); off += (size_t)Hn * Dn * Kn * 2;   // 128 KB
  __bf16* Wqt  = (__bf16*)(ws + off); off += (size_t)Hn * Dn * Kn * 2;
  __bf16* Wvt  = (__bf16*)(ws + off); off += (size_t)Hn * Dn * Dn * 2;   // 1.0 MB
  __bf16* Wot  = (__bf16*)(ws + off); off += (size_t)Hn * Dn * Dn * 2;
  __bf16* Qb   = (__bf16*)(ws + off); off += (size_t)Hn * BS * Kn * 2;   // 1.0 MB
  __bf16* Kb   = (__bf16*)(ws + off); off += (size_t)Hn * BS * Kn * 2;
  __bf16* Kt   = (__bf16*)(ws + off); off += (size_t)Hn * BS * Kn * 2;
  __bf16* Vt   = (__bf16*)(ws + off); off += (size_t)Hn * BS * Dn * 2;   // 8.4 MB
  float*  kcum = (float*)(ws + off);  off += (size_t)Hn * BS * Kn * 4;   // 2.1 MB
  float*  deng = (float*)(ws + off);  off += (size_t)Hn * BS * 4;        // 64 KB
  float*  ckv  = (float*)(ws + off);  off += (size_t)Hn * Bn * NCHUNK * Dn * Kn * 4; // 8.4 MB
  __bf16* ndb  = (__bf16*)(ws + off); off += (size_t)Hn * BS * Dn * 2;   // 8.4 MB

  float* out = (float*)d_out;
  float* att = out + (size_t)Bn * Sn * Dn;

  k_cvt_bf16<<<(BS * Dn + 255) / 256, 256, 0, stream>>>(x, xb, BS * Dn);
  k_cvt_tr<<<(Hn * Dn * Kn + 255) / 256, 256, 0, stream>>>(Wk, Wkt, Dn, Kn, Hn * Dn * Kn);
  k_cvt_tr<<<(Hn * Dn * Kn + 255) / 256, 256, 0, stream>>>(Wq, Wqt, Dn, Kn, Hn * Dn * Kn);
  k_cvt_tr<<<(Hn * Dn * Dn + 255) / 256, 256, 0, stream>>>(Wv, Wvt, Dn, Dn, Hn * Dn * Dn);
  k_cvt_tr<<<(Hn * Dn * Dn + 255) / 256, 256, 0, stream>>>(Wo, Wot, Dn, Dn, Hn * Dn * Dn);

  k_proj_qk<<<Hn * (BS / 16) * 2 / 8, 256, 0, stream>>>(xb, Wkt, Wqt, Qb, Kb, Kt);
  k_proj_vt<<<Hn * (BS / 16) * (Dn / 16) / 8, 256, 0, stream>>>(xb, Wvt, Vt);
  k_kcum<<<Hn * Bn, 32, 0, stream>>>(Kb, kcum);
  k_chunk_kv<<<Hn * Bn * NCHUNK * 32 / 8, 256, 0, stream>>>(Kt, Vt, ckv);
  k_node<<<Hn * Bn * NCHUNK, 256, 0, stream>>>(Qb, Kb, Vt, kcum, ckv, ndb, deng);
  k_att<<<Hn * Bn * 64 * 16 / 8, 256, 0, stream>>>(Qb, Kb, deng, att);
  k_out<<<(BS / 16) * (Dn / 16) / 8, 256, 0, stream>>>(ndb, Wot, out);
}
